// LinearAttention_47914655154475
// MI455X (gfx1250) — matrix operand order repaired and checked by reading, not compiled, not run on hardware
//
#include <hip/hip_runtime.h>
#include <math.h>

// ---------------------------------------------------------------------------
// LinearAttention for MI455X (gfx1250, wave32, WMMA f32_16x16x32_f16)
// ---------------------------------------------------------------------------
typedef __attribute__((ext_vector_type(16))) _Float16 v16h;
typedef __attribute__((ext_vector_type(8)))  _Float16 v8h;
typedef __attribute__((ext_vector_type(8)))  float    v8f;

#define NB    16      // batch
#define NC    128     // channels (== W == LN dim)
#define NH    128
#define NW    128
#define HW    16384   // NH*NW
#define HEADS 8
#define DH    32
#define INNER 256     // HEADS*DH
#define OCH   768     // 3*INNER

// ---- workspace layout (bytes) ----
#define OFF_WQKV16 ((size_t)0)                      // 768*128 f16
#define OFF_WOUT16 ((size_t)196608)                 // 128*256 f16
#define OFF_FM16   ((size_t)262144)                 // 16*128*16384 f16
#define OFF_QKV16  ((size_t)67371008)               // 16*768*16384 f16
#define OFF_KMAX   ((size_t)470024192)              // 4096 f32
#define OFF_KSUM   ((size_t)470040576)              // 4096 f32
#define OFF_CTXT   ((size_t)470056960)              // 128*32*32 f16 (ctx^T)
#define OFF_G16    ((size_t)470319104)              // 16*256*16384 f16
#define OFF_OUT2   OFF_QKV16                        // alias: qkv dead after K5

static __device__ inline v16h cat8(v8h lo, v8h hi) {
  return __builtin_shufflevector(lo, hi, 0,1,2,3,4,5,6,7,8,9,10,11,12,13,14,15);
}
static __device__ inline v8f wmma16(v16h a, v16h b, v8f c) {
  // v_wmma_f32_16x16x32_f16
  return __builtin_amdgcn_wmma_f32_16x16x32_f16(false, a, false, b, (short)0, c,
                                                false, false);
}
// A fragment (16x32 f16): lane L -> row m+(L&15); interleaved K per ISA:
// halves j<8 -> K = kb + kadd + j ; j>=8 -> K = kb + 16 + kadd + (j-8),
// kadd = (L<16) ? 0 : 8.   (base row-major, row stride ld halves)
static __device__ inline v16h load_a(const _Float16* base, int m, int ld, int kb) {
  int lane = threadIdx.x & 31;
  const _Float16* p = base + (size_t)(m + (lane & 15)) * ld + kb + ((lane < 16) ? 0 : 8);
  v8h lo = *(const v8h*)p;
  v8h hi = *(const v8h*)(p + 16);
  return cat8(lo, hi);
}
// B fragment (32x16 f16) from K-transposed storage bt[n][k]:
// lane L -> col n0+(L&15); interleaved K as load_a: j<8 -> kb + kadd + j, j>=8 -> kb + 16 + kadd + (j-8)
static __device__ inline v16h load_b(const _Float16* bt, int n0, int ld, int kb) {
  int lane = threadIdx.x & 31;
  const _Float16* p = bt + (size_t)(n0 + (lane & 15)) * ld + kb + ((lane < 16) ? 0 : 8);
  v8h lo = *(const v8h*)p;
  v8h hi = *(const v8h*)(p + 16);
  return cat8(lo, hi);
}

// ---------------------------------------------------------------------------
// K0: convert weights fp32 -> f16
// ---------------------------------------------------------------------------
__global__ void k_cvt_w(const float* __restrict__ wqkv, const float* __restrict__ wout,
                        _Float16* __restrict__ w16qkv, _Float16* __restrict__ w16out) {
  int i = blockIdx.x * 256 + threadIdx.x;
  if (i < OCH * NC)    w16qkv[i] = (_Float16)wqkv[i];
  if (i < NC * INNER)  w16out[i] = (_Float16)wout[i];
}

// ---------------------------------------------------------------------------
// K1: LayerNorm over W (one wave per 128-float row), write f16
// ---------------------------------------------------------------------------
__global__ void k_ln0(const float* __restrict__ x, const float* __restrict__ g,
                      const float* __restrict__ bt, _Float16* __restrict__ y) {
  int wave = threadIdx.x >> 5, lane = threadIdx.x & 31;
  size_t row = (size_t)blockIdx.x * 8 + wave;      // b*c*h rows
  const float* src = x + row * NW + lane * 4;
  float4 v = *(const float4*)src;
  float xs[4] = {v.x, v.y, v.z, v.w};
  float s = xs[0] + xs[1] + xs[2] + xs[3];
  float q = xs[0]*xs[0] + xs[1]*xs[1] + xs[2]*xs[2] + xs[3]*xs[3];
  #pragma unroll
  for (int m = 16; m >= 1; m >>= 1) {
    s += __shfl_xor(s, m, 32);
    q += __shfl_xor(q, m, 32);
  }
  float mu = s * (1.0f / NW);
  float var = q * (1.0f / NW) - mu * mu;
  float rstd = rsqrtf(var + 1e-5f);
  _Float16* dst = y + row * NW + lane * 4;
  int w = lane * 4;
  #pragma unroll
  for (int j = 0; j < 4; ++j)
    dst[j] = (_Float16)((xs[j] - mu) * rstd * g[w + j] + bt[w + j]);
}

// ---------------------------------------------------------------------------
// K2: qkv GEMM.  per batch: (768 x 128) * (128 x 16384) -> qkv16 (768 x 16384)
// block: 256 thr, tile M=128 N=128 K=128; 8 waves x (16 x 128) strip.
// B fragments batched per K-step so ds_loads overlap the WMMA chain.
// ---------------------------------------------------------------------------
__global__ void k_gemm_qkv(const _Float16* __restrict__ w16,
                           const _Float16* __restrict__ fm16,
                           _Float16* __restrict__ qkv16) {
  __shared__ _Float16 BT[128][136];                // [pos][c], padded
  int nb = blockIdx.x, mb = blockIdx.y, b = blockIdx.z;
  int tid = threadIdx.x;
  size_t posbase = (size_t)nb * 128;
  const _Float16* Bsrc = fm16 + (size_t)b * NC * HW;
  {
    int c = tid >> 1;
    int ph = (tid & 1) * 64;
    const _Float16* src = Bsrc + (size_t)c * HW + posbase + ph;
    #pragma unroll
    for (int i = 0; i < 8; ++i) {
      v8h v = *(const v8h*)(src + i * 8);
      #pragma unroll
      for (int j = 0; j < 8; ++j) BT[ph + i * 8 + j][c] = v[j];
    }
  }
  __syncthreads();
  int wave = tid >> 5;
  int obase = mb * 128 + wave * 16;
  v8f acc[8];
  #pragma unroll
  for (int t = 0; t < 8; ++t) acc[t] = (v8f){0.f,0.f,0.f,0.f,0.f,0.f,0.f,0.f};
  #pragma unroll
  for (int ks = 0; ks < 4; ++ks) {
    v16h af = load_a(w16, obase, NC, ks * 32);
    v16h bf[8];
    #pragma unroll
    for (int nt = 0; nt < 8; ++nt)
      bf[nt] = load_b(&BT[0][0], nt * 16, 136, ks * 32);
    #pragma unroll
    for (int nt = 0; nt < 8; ++nt)
      acc[nt] = wmma16(af, bf[nt], acc[nt]);
  }
  int lane = tid & 31;
  int radd = (lane < 16) ? 0 : 8;
  _Float16* out = qkv16 + (size_t)b * OCH * HW;
  #pragma unroll
  for (int nt = 0; nt < 8; ++nt) {
    size_t pos = posbase + nt * 16 + (lane & 15);
    #pragma unroll
    for (int r = 0; r < 8; ++r)
      out[(size_t)(obase + r + radd) * HW + pos] = (_Float16)acc[nt][r];
  }
}

// ---------------------------------------------------------------------------
// K3: column max / sum(exp) over n=16384 for k-softmax. 1 block per column.
// col = b*256 + (head*32+d);  channel = 256 + col%256
// ---------------------------------------------------------------------------
__global__ void k_kstats(const _Float16* __restrict__ qkv16,
                         float* __restrict__ kmax, float* __restrict__ ksum) {
  __shared__ float red[256];
  int col = blockIdx.x;
  int b = col >> 8, kc = col & 255;
  const _Float16* p = qkv16 + ((size_t)b * OCH + 256 + kc) * HW;
  int tid = threadIdx.x;
  float m = -INFINITY;
  for (int i = tid; i < HW; i += 256) {
    __builtin_prefetch(p + i + 2048, 0, 0);        // global_prefetch_b8
    m = fmaxf(m, (float)p[i]);
  }
  red[tid] = m; __syncthreads();
  for (int s = 128; s > 0; s >>= 1) {
    if (tid < s) red[tid] = fmaxf(red[tid], red[tid + s]);
    __syncthreads();
  }
  m = red[0]; __syncthreads();
  float s = 0.f;
  for (int i = tid; i < HW; i += 256) s += expf((float)p[i] - m);
  red[tid] = s; __syncthreads();
  for (int st = 128; st > 0; st >>= 1) {
    if (tid < st) red[tid] += red[tid + st];
    __syncthreads();
  }
  if (tid == 0) { kmax[col] = m; ksum[col] = red[0]; }
}

// ---------------------------------------------------------------------------
// K4: context[bh] = (1/n) * sum_n softmax_n(k)[n,d] * l2norm(v)[n,e]
// 1 block per bh; VALU FMAs (4.3 GFLOP total - negligible). writes ctx^T f16
// ---------------------------------------------------------------------------
__global__ void k_context(const _Float16* __restrict__ qkv16,
                          const float* __restrict__ kmax, const float* __restrict__ ksum,
                          _Float16* __restrict__ ctxT) {
  __shared__ float   smax[32], sinv[32];
  __shared__ _Float16 lk[32][64];     // [d][nn]
  __shared__ _Float16 lv[32][64];     // [e][nn]
  __shared__ float   kf[32][65];      // exp(k - max)
  __shared__ float   vf[32][65];      // v / ||v||
  __shared__ float   vw[64];
  int bh = blockIdx.x;
  int b = bh >> 3, head = bh & 7;
  int tid = threadIdx.x;
  if (tid < 32) {
    int col = bh * 32 + tid;
    smax[tid] = kmax[col];
    sinv[tid] = 1.0f / ksum[col];
  }
  int d0 = tid >> 3;                  // 0..31 (also load channel)
  int e0 = (tid & 7) * 4;             // 0..28
  int seg = (tid & 7) * 8;            // 0..56
  const _Float16* kp = qkv16 + ((size_t)b * OCH + 256 + head * 32 + d0) * HW + seg;
  const _Float16* vp = qkv16 + ((size_t)b * OCH + 512 + head * 32 + d0) * HW + seg;
  float a0 = 0.f, a1 = 0.f, a2 = 0.f, a3 = 0.f;
  __syncthreads();
  for (int n0 = 0; n0 < HW; n0 += 64) {
    v8h kv = *(const v8h*)(kp + n0);
    v8h vv = *(const v8h*)(vp + n0);
    __builtin_prefetch(kp + n0 + 512, 0, 0);
    __builtin_prefetch(vp + n0 + 512, 0, 0);
    #pragma unroll
    for (int j = 0; j < 8; ++j) { lk[d0][seg + j] = kv[j]; lv[d0][seg + j] = vv[j]; }
    __syncthreads();
    {
      float m = smax[d0];
      #pragma unroll
      for (int j = 0; j < 8; ++j)
        kf[d0][seg + j] = expf((float)lk[d0][seg + j] - m);
    }
    if (tid < 64) {
      float s = 0.f;
      #pragma unroll
      for (int e = 0; e < 32; ++e) { float x = (float)lv[e][tid]; s += x * x; }
      vw[tid] = 1.0f / fmaxf(sqrtf(s), 1e-12f);
    }
    __syncthreads();
    #pragma unroll
    for (int j = 0; j < 8; ++j)
      vf[d0][seg + j] = (float)lv[d0][seg + j] * vw[seg + j];
    __syncthreads();
    #pragma unroll 4
    for (int nn = 0; nn < 64; ++nn) {
      float kk = kf[d0][nn];
      a0 = fmaf(kk, vf[e0 + 0][nn], a0);
      a1 = fmaf(kk, vf[e0 + 1][nn], a1);
      a2 = fmaf(kk, vf[e0 + 2][nn], a2);
      a3 = fmaf(kk, vf[e0 + 3][nn], a3);
    }
    __syncthreads();
  }
  float sc = sinv[d0] * (1.0f / 16384.0f);
  _Float16* dst = ctxT + (size_t)bh * 1024;       // [e][d]
  dst[(e0 + 0) * 32 + d0] = (_Float16)(a0 * sc);
  dst[(e0 + 1) * 32 + d0] = (_Float16)(a1 * sc);
  dst[(e0 + 2) * 32 + d0] = (_Float16)(a2 * sc);
  dst[(e0 + 3) * 32 + d0] = (_Float16)(a3 * sc);
}

// ---------------------------------------------------------------------------
// K5: out = softmax_d(q)*scale  x  context  (K=32 -> one WMMA per tile),
//     then GELU(erf), store g f16 in (b, inner, pos) layout.
// block: (bh, 128-pos chunk); 8 waves x one 16-n tile x 2 e-tiles
// ---------------------------------------------------------------------------
__global__ void k_attn_out(const _Float16* __restrict__ qkv16,
                           const _Float16* __restrict__ ctxT,
                           _Float16* __restrict__ g16) {
  __shared__ _Float16 Q[128][40];     // [n][d]
  __shared__ _Float16 CT[32][40];     // [e][d]
  __shared__ float    O[128][34];     // [n][e]
  int nb = blockIdx.x, bh = blockIdx.y;
  int b = bh >> 3, head = bh & 7;
  int tid = threadIdx.x;
  size_t nbase = (size_t)nb * 128;
  {
    #pragma unroll
    for (int i = 0; i < 4; ++i) {
      int idx = tid * 4 + i;          // 0..1023
      CT[idx >> 5][idx & 31] = ctxT[(size_t)bh * 1024 + idx];
    }
  }
  {
    int d = tid & 31;
    int nseg = (tid >> 5) * 16;
    const _Float16* src = qkv16 + ((size_t)b * OCH + head * 32 + d) * HW + nbase + nseg;
    v8h q0 = *(const v8h*)src;
    v8h q1 = *(const v8h*)(src + 8);
    #pragma unroll
    for (int j = 0; j < 8; ++j) { Q[nseg + j][d] = q0[j]; Q[nseg + 8 + j][d] = q1[j]; }
  }
  __syncthreads();
  if (tid < 128) {                    // row softmax over d=32, * 1/sqrt(32)
    float vals[32];
    float m = -INFINITY;
    #pragma unroll
    for (int d = 0; d < 32; ++d) { vals[d] = (float)Q[tid][d]; m = fmaxf(m, vals[d]); }
    float s = 0.f;
    #pragma unroll
    for (int d = 0; d < 32; ++d) { vals[d] = expf(vals[d] - m); s += vals[d]; }
    float inv = 0.17677669529663687f / s;   // scale / sum
    #pragma unroll
    for (int d = 0; d < 32; ++d) Q[tid][d] = (_Float16)(vals[d] * inv);
  }
  __syncthreads();
  int wave = tid >> 5, lane = tid & 31;
  v16h af = load_a(&Q[0][0], wave * 16, 40, 0);
  #pragma unroll
  for (int et = 0; et < 2; ++et) {
    v16h bf = load_b(&CT[0][0], et * 16, 40, 0);
    v8f c = (v8f){0.f,0.f,0.f,0.f,0.f,0.f,0.f,0.f};
    c = wmma16(af, bf, c);
    int e = et * 16 + (lane & 15);
    int radd = (lane < 16) ? 0 : 8;
    #pragma unroll
    for (int r = 0; r < 8; ++r) O[wave * 16 + r + radd][e] = c[r];
  }
  __syncthreads();
  {
    int e = tid >> 3;
    int nseg = (tid & 7) * 16;
    _Float16* dst = g16 + ((size_t)b * INNER + head * 32 + e) * HW + nbase + nseg;
    #pragma unroll
    for (int i = 0; i < 16; ++i) {
      float x = O[nseg + i][e];
      float ge = 0.5f * x * (1.0f + erff(x * 0.7071067811865476f)); // exact GELU
      dst[i] = (_Float16)ge;
    }
  }
}

// ---------------------------------------------------------------------------
// K6: out2 = w_out (128x256) x g (256x16384) per batch -> f32
// block tile M=128 N=64 K=256; 8 waves x (16 x 64) strip
// ---------------------------------------------------------------------------
__global__ void k_gemm_out(const _Float16* __restrict__ w16,
                           const _Float16* __restrict__ g16,
                           float* __restrict__ out2) {
  __shared__ _Float16 BT[64][264];                 // [pos][i], padded
  int nb = blockIdx.x, b = blockIdx.y;
  int tid = threadIdx.x;
  size_t posbase = (size_t)nb * 64;
  {
    int c = tid;                                   // 0..255
    const _Float16* src = g16 + ((size_t)b * INNER + c) * HW + posbase;
    #pragma unroll
    for (int i = 0; i < 8; ++i) {
      v8h v = *(const v8h*)(src + i * 8);
      #pragma unroll
      for (int j = 0; j < 8; ++j) BT[i * 8 + j][c] = v[j];
    }
  }
  __syncthreads();
  int wave = tid >> 5;
  int obase = wave * 16;
  v8f acc[4];
  #pragma unroll
  for (int t = 0; t < 4; ++t) acc[t] = (v8f){0.f,0.f,0.f,0.f,0.f,0.f,0.f,0.f};
  #pragma unroll
  for (int ks = 0; ks < 8; ++ks) {
    v16h af = load_a(w16, obase, INNER, ks * 32);
    v16h bf[4];
    #pragma unroll
    for (int nt = 0; nt < 4; ++nt)
      bf[nt] = load_b(&BT[0][0], nt * 16, 264, ks * 32);
    #pragma unroll
    for (int nt = 0; nt < 4; ++nt)
      acc[nt] = wmma16(af, bf[nt], acc[nt]);
  }
  int lane = tid & 31;
  int radd = (lane < 16) ? 0 : 8;
  float* out = out2 + (size_t)b * NC * HW;
  #pragma unroll
  for (int nt = 0; nt < 4; ++nt) {
    size_t pos = posbase + nt * 16 + (lane & 15);
    #pragma unroll
    for (int r = 0; r < 8; ++r)
      out[(size_t)(obase + r + radd) * HW + pos] = acc[nt][r];
  }
}

// ---------------------------------------------------------------------------
// K7: final LayerNorm over W (f32 -> f32 into d_out)
// ---------------------------------------------------------------------------
__global__ void k_ln1(const float* __restrict__ x, const float* __restrict__ g,
                      const float* __restrict__ bt, float* __restrict__ y) {
  int wave = threadIdx.x >> 5, lane = threadIdx.x & 31;
  size_t row = (size_t)blockIdx.x * 8 + wave;
  const float* src = x + row * NW + lane * 4;
  float4 v = *(const float4*)src;
  float xs[4] = {v.x, v.y, v.z, v.w};
  float s = xs[0] + xs[1] + xs[2] + xs[3];
  float q = xs[0]*xs[0] + xs[1]*xs[1] + xs[2]*xs[2] + xs[3]*xs[3];
  #pragma unroll
  for (int m = 16; m >= 1; m >>= 1) {
    s += __shfl_xor(s, m, 32);
    q += __shfl_xor(q, m, 32);
  }
  float mu = s * (1.0f / NW);
  float var = q * (1.0f / NW) - mu * mu;
  float rstd = rsqrtf(var + 1e-5f);
  float* dst = y + row * NW + lane * 4;
  int w = lane * 4;
  #pragma unroll
  for (int j = 0; j < 4; ++j)
    dst[j] = (xs[j] - mu) * rstd * g[w + j] + bt[w + j];
}

// ---------------------------------------------------------------------------
extern "C" void kernel_launch(void* const* d_in, const int* in_sizes, int n_in,
                              void* d_out, int out_size, void* d_ws, size_t ws_size,
                              hipStream_t stream) {
  (void)in_sizes; (void)n_in; (void)out_size; (void)ws_size;
  const float* fmap = (const float*)d_in[0];
  const float* ln0g = (const float*)d_in[1];
  const float* ln0b = (const float*)d_in[2];
  const float* wqkv = (const float*)d_in[3];
  const float* wout = (const float*)d_in[4];
  const float* ln1g = (const float*)d_in[5];
  const float* ln1b = (const float*)d_in[6];
  float* out = (float*)d_out;
  char* ws = (char*)d_ws;

  _Float16* w16qkv = (_Float16*)(ws + OFF_WQKV16);
  _Float16* w16out = (_Float16*)(ws + OFF_WOUT16);
  _Float16* fm16   = (_Float16*)(ws + OFF_FM16);
  _Float16* qkv16  = (_Float16*)(ws + OFF_QKV16);
  float*    kmax   = (float*)   (ws + OFF_KMAX);
  float*    ksum   = (float*)   (ws + OFF_KSUM);
  _Float16* ctxT   = (_Float16*)(ws + OFF_CTXT);
  _Float16* g16    = (_Float16*)(ws + OFF_G16);
  float*    out2   = (float*)   (ws + OFF_OUT2);

  k_cvt_w   <<<dim3(384),          dim3(256), 0, stream>>>(wqkv, wout, w16qkv, w16out);
  k_ln0     <<<dim3(32768),        dim3(256), 0, stream>>>(fmap, ln0g, ln0b, fm16);
  k_gemm_qkv<<<dim3(128, 6, 16),   dim3(256), 0, stream>>>(w16qkv, fm16, qkv16);
  k_kstats  <<<dim3(4096),         dim3(256), 0, stream>>>(qkv16, kmax, ksum);
  k_context <<<dim3(128),          dim3(256), 0, stream>>>(qkv16, kmax, ksum, ctxT);
  k_attn_out<<<dim3(128, 128),     dim3(256), 0, stream>>>(qkv16, ctxT, g16);
  k_gemm_out<<<dim3(256, 16),      dim3(256), 0, stream>>>(w16out, g16, out2);
  k_ln1     <<<dim3(32768),        dim3(256), 0, stream>>>(out2, ln1g, ln1b, out);
}
